// CurvePredictor_8744553415171
// MI455X (gfx1250) — matrix operand order repaired and checked by reading, not compiled, not run on hardware
//
#include <hip/hip_runtime.h>
#include <cstdint>
#include <cstddef>

// ---------------------------------------------------------------------------
// CurvePredictor on MI455X (gfx1250): encoder -> 2-layer LSTM (T=256) -> decoder
// All heavy math on v_wmma_f32_16x16x32_f16 (wave32 WMMA), f32 accumulate.
// ---------------------------------------------------------------------------

#define BB 2048
#define TT 256
#define HH 256
#define LEPS 1e-5f

typedef __attribute__((ext_vector_type(16))) _Float16 v16h;
typedef __attribute__((ext_vector_type(8)))  float    v8f;
typedef __attribute__((ext_vector_type(4)))  float    v4f;

union FragH { v16h v; v4f f[2]; };   // 32B: one WMMA A/B operand per lane

__device__ __forceinline__ float gelu_f(float x) {
  return 0.5f * x * (1.0f + erff(x * 0.70710678118654752f));
}
__device__ __forceinline__ float sigm_f(float x) {
  return 1.0f / (1.0f + expf(-x));
}

// ---------------------------------------------------------------------------
// utility kernels
// ---------------------------------------------------------------------------
__global__ void k_zero(v4f* __restrict__ p, int n) {
  int i = blockIdx.x * blockDim.x + threadIdx.x;
  if (i < n) { v4f z = {0.f, 0.f, 0.f, 0.f}; p[i] = z; }
}

__global__ void k_cvt(const float* __restrict__ s, _Float16* __restrict__ d, int n) {
  int i = blockIdx.x * blockDim.x + threadIdx.x;
  if (i < n) d[i] = (_Float16)s[i];
}

__global__ void k_addb(const float* __restrict__ a, const float* __restrict__ b,
                       float* __restrict__ o, int n) {
  int i = blockIdx.x * blockDim.x + threadIdx.x;
  if (i < n) o[i] = a[i] + b[i];
}

// ---------------------------------------------------------------------------
// encoder: one block per batch row. Linear(5->128)+GELU+LN, Linear(128->256)+GELU+LN
// Writes enc in f16 (WMMA operand for the xp0 GEMM).
// ---------------------------------------------------------------------------
__global__ __launch_bounds__(256) void k_encoder(
    const float* __restrict__ x,
    const float* __restrict__ W1, const float* __restrict__ b1,
    const float* __restrict__ g1, const float* __restrict__ be1,
    const float* __restrict__ W2, const float* __restrict__ b2,
    const float* __restrict__ g2, const float* __restrict__ be2,
    _Float16* __restrict__ enc16)
{
  const int b = blockIdx.x;
  const int tid = threadIdx.x;
  __shared__ float xs[5];
  __shared__ float v1[128];
  __shared__ float rs[256], rq[256];

  if (tid < 5) xs[tid] = x[(size_t)b * 5 + tid];
  __syncthreads();

  float a = 0.f;
  if (tid < 128) {
    float s = b1[tid];
#pragma unroll
    for (int j = 0; j < 5; ++j) s += xs[j] * W1[tid * 5 + j];
    a = gelu_f(s);
  }
  rs[tid] = (tid < 128) ? a : 0.f;
  rq[tid] = (tid < 128) ? a * a : 0.f;
  __syncthreads();
  for (int s = 128; s > 0; s >>= 1) {
    if (tid < s) { rs[tid] += rs[tid + s]; rq[tid] += rq[tid + s]; }
    __syncthreads();
  }
  float mean = rs[0] * (1.f / 128.f);
  float var  = rq[0] * (1.f / 128.f) - mean * mean;
  if (tid < 128) v1[tid] = (a - mean) * rsqrtf(var + LEPS) * g1[tid] + be1[tid];
  __syncthreads();

  float s2 = b2[tid];
  for (int k = 0; k < 128; ++k) s2 += v1[k] * W2[tid * 128 + k];
  float a2 = gelu_f(s2);
  rs[tid] = a2; rq[tid] = a2 * a2;
  __syncthreads();
  for (int s = 128; s > 0; s >>= 1) {
    if (tid < s) { rs[tid] += rs[tid + s]; rq[tid] += rq[tid + s]; }
    __syncthreads();
  }
  float mean2 = rs[0] * (1.f / 256.f);
  float var2  = rq[0] * (1.f / 256.f) - mean2 * mean2;
  enc16[(size_t)b * 256 + tid] =
      (_Float16)((a2 - mean2) * rsqrtf(var2 + LEPS) * g2[tid] + be2[tid]);
}

// ---------------------------------------------------------------------------
// Fused WMMA GEMM + LSTM cell.
//   gates[b, g*256+hc] = sum_phases A@Wᵀ  (+ Cin)
// Block tile: 64 batch rows x 32 h-cols; the 4 gate chunks for that tile are
// computed by the same block so i/f/g/o + c/h update happen in registers.
// MODE 0: plain GEMM out (xp0 = enc@Wih0ᵀ + bias), Cin = bias[1024], Cst = out[M,1024]
// MODE 1: layer-0 cell, Cin = xp0[M,1024] (per-element), A0=h0_prev, W0=Whh0
// MODE 2: layer-1 cell, two K-phases: (h0_t, Wih1) + (h1_prev, Whh1), Cin = bias1
// ---------------------------------------------------------------------------
template <int MODE>
__global__ __launch_bounds__(256) void k_gemm_cell(
    const _Float16* __restrict__ A0, const _Float16* __restrict__ W0,
    const _Float16* __restrict__ A1, const _Float16* __restrict__ W1,
    const float* __restrict__ Cin, float* __restrict__ Cst,
    _Float16* __restrict__ Hout)
{
  __shared__ __align__(16) _Float16 Asm[64 * 32];    // A tile (rows x K32)
  __shared__ __align__(16) _Float16 Wsm[128 * 32];   // 4 gates x 32 n-rows x K32

  const int tid  = threadIdx.x;
  const int lane = tid & 31;
  const int wave = tid >> 5;
  const int mi = wave >> 1;        // 0..3 : 16-row tile within 64
  const int ni = wave & 1;         // 0..1 : 16-col tile within 32
  const int mBase  = blockIdx.x * 64;
  const int hcBase = blockIdx.y * 32;

  v8f acc[4];
#pragma unroll
  for (int g = 0; g < 4; ++g)
#pragma unroll
    for (int i = 0; i < 8; ++i) acc[g][i] = 0.f;

  const int lrow = tid >> 2, lcg = tid & 3;   // A stager: 64 rows x 4 x 8h
  const int wr = tid >> 1, wpart = tid & 1;   // W stager: 128 rows x 2 x 16h

  const int NPHASE = (MODE == 2) ? 2 : 1;
  for (int ph = 0; ph < NPHASE; ++ph) {
    const _Float16* Ap = (MODE == 2 && ph == 1) ? A1 : A0;
    const _Float16* Wp = (MODE == 2 && ph == 1) ? W1 : W0;
    for (int k0 = 0; k0 < 256; k0 += 32) {
      *reinterpret_cast<v4f*>(&Asm[lrow * 32 + lcg * 8]) =
          *reinterpret_cast<const v4f*>(Ap + (size_t)(mBase + lrow) * 256 + k0 + lcg * 8);
      {
        const int n = (wr >> 5) * 256 + hcBase + (wr & 31);   // gate*256 + hcol
        const _Float16* s = Wp + (size_t)n * 256 + k0 + wpart * 16;
        *reinterpret_cast<v4f*>(&Wsm[wr * 32 + wpart * 16])     = *reinterpret_cast<const v4f*>(s);
        *reinterpret_cast<v4f*>(&Wsm[wr * 32 + wpart * 16 + 8]) = *reinterpret_cast<const v4f*>(s + 8);
      }
      __syncthreads();

      // A fragment: lanes 0-15 hold K {0..7,16..23}, lanes 16-31 hold {8..15,24..31}
      FragH a;
      const int arow = mi * 16 + (lane & 15);
      const int kA = (lane & 16) ? 8 : 0;
      a.f[0] = *reinterpret_cast<const v4f*>(&Asm[arow * 32 + kA]);
      a.f[1] = *reinterpret_cast<const v4f*>(&Asm[arow * 32 + kA + 16]);

      // B fragment: lanes 0-15 hold K {0..7,16..23}, lanes 16-31 hold {8..15,24..31}
      const int kB = (lane & 16) ? 8 : 0;
#pragma unroll
      for (int g = 0; g < 4; ++g) {
        FragH bf;
        const int br = g * 32 + ni * 16 + (lane & 15);
        bf.f[0] = *reinterpret_cast<const v4f*>(&Wsm[br * 32 + kB]);
        bf.f[1] = *reinterpret_cast<const v4f*>(&Wsm[br * 32 + kB + 16]);
        acc[g] = __builtin_amdgcn_wmma_f32_16x16x32_f16(
            false, a.v, false, bf.v, (short)0, acc[g], false, false);
      }
      __syncthreads();
    }
  }

  // epilogue: C/D layout -> M = i + 8*(lane>=16), N = lane&15
  const int moff = (lane & 16) ? 8 : 0;
  const int hidx = hcBase + ni * 16 + (lane & 15);
#pragma unroll
  for (int i = 0; i < 8; ++i) {
    const int bb = mBase + mi * 16 + moff + i;
    float gv[4];
#pragma unroll
    for (int g = 0; g < 4; ++g) {
      float cadd;
      if constexpr (MODE == 1) cadd = Cin[(size_t)bb * 1024 + g * 256 + hidx];
      else                     cadd = Cin[g * 256 + hidx];
      gv[g] = acc[g][i] + cadd;
    }
    if constexpr (MODE == 0) {
#pragma unroll
      for (int g = 0; g < 4; ++g)
        Cst[(size_t)bb * 1024 + g * 256 + hidx] = gv[g];
    } else {
      const float ig = sigm_f(gv[0]);       // torch gate order i,f,g,o
      const float fg = sigm_f(gv[1]);
      const float gg = tanhf(gv[2]);
      const float og = sigm_f(gv[3]);
      const size_t ci = (size_t)bb * 256 + hidx;
      const float cn = fg * Cst[ci] + ig * gg;
      Cst[ci] = cn;
      Hout[ci] = (_Float16)(og * tanhf(cn));
    }
  }
}

// ---------------------------------------------------------------------------
// Decoder for one timestep: d = gelu(h1 @ W3ᵀ + b3); out[:,t,:] = d @ W4ᵀ + b4
// Block tile: 64 rows x all 128 d-cols (so the 128->2 projection fuses in-block).
// ---------------------------------------------------------------------------
__global__ __launch_bounds__(256) void k_decoder(
    const _Float16* __restrict__ Hcur, const _Float16* __restrict__ W3h,
    const float* __restrict__ b3, const float* __restrict__ W4,
    const float* __restrict__ b4, float* __restrict__ out, int t)
{
  __shared__ __align__(16) _Float16 Asm[64 * 32];
  __shared__ __align__(16) _Float16 Wsm[128 * 32];
  __shared__ float Dt[64 * 128];
  __shared__ float red[512];

  const int tid  = threadIdx.x;
  const int lane = tid & 31;
  const int wave = tid >> 5;
  const int mi = wave >> 1;       // 0..3
  const int nh = wave & 1;        // 0..1 (which 64-col half)
  const int mBase = blockIdx.x * 64;

  v8f acc[4];
#pragma unroll
  for (int g = 0; g < 4; ++g)
#pragma unroll
    for (int i = 0; i < 8; ++i) acc[g][i] = 0.f;

  const int lrow = tid >> 2, lcg = tid & 3;
  const int wr = tid >> 1, wpart = tid & 1;

  for (int k0 = 0; k0 < 256; k0 += 32) {
    *reinterpret_cast<v4f*>(&Asm[lrow * 32 + lcg * 8]) =
        *reinterpret_cast<const v4f*>(Hcur + (size_t)(mBase + lrow) * 256 + k0 + lcg * 8);
    {
      const _Float16* s = W3h + (size_t)wr * 256 + k0 + wpart * 16;
      *reinterpret_cast<v4f*>(&Wsm[wr * 32 + wpart * 16])     = *reinterpret_cast<const v4f*>(s);
      *reinterpret_cast<v4f*>(&Wsm[wr * 32 + wpart * 16 + 8]) = *reinterpret_cast<const v4f*>(s + 8);
    }
    __syncthreads();

    FragH a;
    const int arow = mi * 16 + (lane & 15);
    const int kA = (lane & 16) ? 8 : 0;
    a.f[0] = *reinterpret_cast<const v4f*>(&Asm[arow * 32 + kA]);
    a.f[1] = *reinterpret_cast<const v4f*>(&Asm[arow * 32 + kA + 16]);

    const int kB = (lane & 16) ? 8 : 0;
#pragma unroll
    for (int nj = 0; nj < 4; ++nj) {
      FragH bf;
      const int br = nh * 64 + nj * 16 + (lane & 15);
      bf.f[0] = *reinterpret_cast<const v4f*>(&Wsm[br * 32 + kB]);
      bf.f[1] = *reinterpret_cast<const v4f*>(&Wsm[br * 32 + kB + 16]);
      acc[nj] = __builtin_amdgcn_wmma_f32_16x16x32_f16(
          false, a.v, false, bf.v, (short)0, acc[nj], false, false);
    }
    __syncthreads();
  }

  const int moff = (lane & 16) ? 8 : 0;
#pragma unroll
  for (int nj = 0; nj < 4; ++nj) {
    const int n = nh * 64 + nj * 16 + (lane & 15);
#pragma unroll
    for (int i = 0; i < 8; ++i) {
      const int m = mi * 16 + moff + i;
      Dt[m * 128 + n] = gelu_f(acc[nj][i] + b3[n]);
    }
  }
  __syncthreads();

  // 128 -> 2 projection: 4 segment-partials per row, LDS reduce
  const int row = tid >> 2, seg = tid & 3;
  float p0 = 0.f, p1 = 0.f;
  for (int k = seg * 32; k < seg * 32 + 32; ++k) {
    const float dv = Dt[row * 128 + k];
    p0 += dv * W4[k];
    p1 += dv * W4[128 + k];
  }
  red[tid] = p0; red[256 + tid] = p1;
  __syncthreads();
  if (seg == 0) {
    float o0 = b4[0], o1 = b4[1];
#pragma unroll
    for (int s = 0; s < 4; ++s) { o0 += red[tid + s]; o1 += red[256 + tid + s]; }
    const size_t ob = (size_t)(mBase + row) * (TT * 2) + (size_t)t * 2;
    out[ob] = o0; out[ob + 1] = o1;
  }
}

// ---------------------------------------------------------------------------
// host orchestration
// ---------------------------------------------------------------------------
extern "C" void kernel_launch(void* const* d_in, const int* in_sizes, int n_in,
                              void* d_out, int out_size, void* d_ws, size_t ws_size,
                              hipStream_t stream)
{
  (void)in_sizes; (void)n_in; (void)out_size; (void)ws_size;
  const float* x    = (const float*)d_in[0];
  const float* W1   = (const float*)d_in[1];
  const float* b1   = (const float*)d_in[2];
  const float* g1   = (const float*)d_in[3];
  const float* be1  = (const float*)d_in[4];
  const float* W2   = (const float*)d_in[5];
  const float* b2   = (const float*)d_in[6];
  const float* g2   = (const float*)d_in[7];
  const float* be2  = (const float*)d_in[8];
  const float* Wih0 = (const float*)d_in[9];
  const float* Whh0 = (const float*)d_in[10];
  const float* bih0 = (const float*)d_in[11];
  const float* bhh0 = (const float*)d_in[12];
  const float* Wih1 = (const float*)d_in[13];
  const float* Whh1 = (const float*)d_in[14];
  const float* bih1 = (const float*)d_in[15];
  const float* bhh1 = (const float*)d_in[16];
  const float* W3   = (const float*)d_in[17];
  const float* b3   = (const float*)d_in[18];
  const float* W4   = (const float*)d_in[19];
  const float* b4   = (const float*)d_in[20];
  float* out = (float*)d_out;

  uint8_t* ws = (uint8_t*)d_ws;
  size_t off = 0;
  auto alloc = [&](size_t bytes) -> void* {
    void* p = ws + off;
    off += bytes;
    off = (off + 255) & ~(size_t)255;
    return p;
  };

  _Float16* enc16 = (_Float16*)alloc((size_t)BB * HH * 2);
  _Float16* Wih0h = (_Float16*)alloc((size_t)1024 * 256 * 2);
  _Float16* Whh0h = (_Float16*)alloc((size_t)1024 * 256 * 2);
  _Float16* Wih1h = (_Float16*)alloc((size_t)1024 * 256 * 2);
  _Float16* Whh1h = (_Float16*)alloc((size_t)1024 * 256 * 2);
  _Float16* W3h   = (_Float16*)alloc((size_t)128 * 256 * 2);
  float* bias0 = (float*)alloc(1024 * 4);
  float* bias1 = (float*)alloc(1024 * 4);
  float* xp0   = (float*)alloc((size_t)BB * 1024 * 4);
  _Float16* h0buf[2] = { (_Float16*)alloc((size_t)BB * HH * 2),
                         (_Float16*)alloc((size_t)BB * HH * 2) };
  _Float16* h1buf[2] = { (_Float16*)alloc((size_t)BB * HH * 2),
                         (_Float16*)alloc((size_t)BB * HH * 2) };
  float* c0 = (float*)alloc((size_t)BB * HH * 4);
  float* c1 = (float*)alloc((size_t)BB * HH * 4);

  // --- state init (h0, c0 start at zero every call) ---
  {
    const int nh16 = BB * HH * 2 / 16;   // f16 buffer in v4f units
    const int nf32 = BB * HH * 4 / 16;   // f32 buffer in v4f units
    k_zero<<<(nh16 + 255) / 256, 256, 0, stream>>>((v4f*)h0buf[0], nh16);
    k_zero<<<(nh16 + 255) / 256, 256, 0, stream>>>((v4f*)h1buf[0], nh16);
    k_zero<<<(nf32 + 255) / 256, 256, 0, stream>>>((v4f*)c0, nf32);
    k_zero<<<(nf32 + 255) / 256, 256, 0, stream>>>((v4f*)c1, nf32);
  }

  // --- weight conversion to f16 + bias combine ---
  {
    const int nw = 1024 * 256;
    k_cvt<<<(nw + 255) / 256, 256, 0, stream>>>(Wih0, Wih0h, nw);
    k_cvt<<<(nw + 255) / 256, 256, 0, stream>>>(Whh0, Whh0h, nw);
    k_cvt<<<(nw + 255) / 256, 256, 0, stream>>>(Wih1, Wih1h, nw);
    k_cvt<<<(nw + 255) / 256, 256, 0, stream>>>(Whh1, Whh1h, nw);
    const int nw3 = 128 * 256;
    k_cvt<<<(nw3 + 255) / 256, 256, 0, stream>>>(W3, W3h, nw3);
    k_addb<<<4, 256, 0, stream>>>(bih0, bhh0, bias0, 1024);
    k_addb<<<4, 256, 0, stream>>>(bih1, bhh1, bias1, 1024);
  }

  // --- encoder ---
  k_encoder<<<BB, 256, 0, stream>>>(x, W1, b1, g1, be1, W2, b2, g2, be2, enc16);

  // --- xp0 = enc @ Wih0ᵀ + (bih0 + bhh0), computed once ---
  k_gemm_cell<0><<<dim3(32, 8), 256, 0, stream>>>(
      enc16, Wih0h, (const _Float16*)nullptr, (const _Float16*)nullptr,
      bias0, xp0, (_Float16*)nullptr);

  // --- interleaved recurrence + decoder per step ---
  for (int t = 0; t < TT; ++t) {
    const _Float16* h0p = h0buf[t & 1];
    _Float16*       h0n = h0buf[(t + 1) & 1];
    const _Float16* h1p = h1buf[t & 1];
    _Float16*       h1n = h1buf[(t + 1) & 1];

    // layer 0: gates = xp0 + h0 @ Whh0ᵀ -> cell
    k_gemm_cell<1><<<dim3(32, 8), 256, 0, stream>>>(
        h0p, Whh0h, (const _Float16*)nullptr, (const _Float16*)nullptr,
        xp0, c0, h0n);

    // layer 1: gates = h0_t @ Wih1ᵀ + h1 @ Whh1ᵀ + (bih1 + bhh1) -> cell
    k_gemm_cell<2><<<dim3(32, 8), 256, 0, stream>>>(
        h0n, Wih1h, h1p, Whh1h, bias1, c1, h1n);

    // decoder for this step -> out[:, t, :]
    k_decoder<<<32, 256, 0, stream>>>(h1n, W3h, b3, W4, b4, out, t);
  }
}
